// RecursiveTreeGNN_37864431681857
// MI455X (gfx1250) — compile-verified, matrix operand order checked by reading, not run on hardware
//
#include <hip/hip_runtime.h>
#include <hip/hip_bf16.h>
#include <stdint.h>

// ---------------------------------------------------------------------------
// Recursive Tree-LSTM GNN on MI455X (gfx1250), bf16 WMMA path.
// B_ARY=4, DEPTH=8, H=512, IN=512, OUT=256, L=2, N=87381.
// GEMM: 256-thread WG = 8 waves; wave tile 32(M)x64(N); B tile staged in LDS
// (double-buffered) via async global->LDS when available.
// ---------------------------------------------------------------------------

#define HF      512
#define OUTF    256
#define NNODES  87381
#define NPAD    87552          // NNODES rounded up to 256 rows
#define TDEPTH  8
#define CHUNK   4096
#define LDS_STRIDE 40          // bf16 elements per LDS B-tile row (32 used + pad)

typedef __attribute__((ext_vector_type(16))) __bf16 v16bf;
typedef __attribute__((ext_vector_type(8)))  float  v8f;
typedef int v4i_ __attribute__((vector_size(16)));   // matches builtin param type

union FragBF { v16bf v; uint4 q[2]; };
union AccF   { v8f   v; float f[8]; };

#ifdef __has_builtin
#if __has_builtin(__builtin_amdgcn_global_load_async_to_lds_b128)
#define HAVE_ASYNC_LDS 1
#endif
#endif
#ifndef HAVE_ASYNC_LDS
#define HAVE_ASYNC_LDS 0
#endif

__device__ __forceinline__ unsigned short f2bf(float f) {
  unsigned u = __float_as_uint(f);
  u += 0x7FFFu + ((u >> 16) & 1u);          // round-to-nearest-even
  return (unsigned short)(u >> 16);
}
__device__ __forceinline__ float bf2f(unsigned short b) {
  return __uint_as_float(((unsigned)b) << 16);
}
__device__ __forceinline__ float sigmoidf_(float x) {
  return 1.0f / (1.0f + expf(-x));
}

// Stage one 16-byte chunk of the B tile into LDS.
__device__ __forceinline__ void stage_b128(const unsigned short* g,
                                           unsigned short* s) {
#if HAVE_ASYNC_LDS
  __builtin_amdgcn_global_load_async_to_lds_b128((v4i_*)g, (v4i_*)s,
                                                 /*imm offset*/0, /*cpol*/0);
#else
  *(uint4*)s = *(const uint4*)g;
#endif
}

__device__ __forceinline__ void async_wait0() {
#if HAVE_ASYNC_LDS
#if __has_builtin(__builtin_amdgcn_s_wait_asynccnt)
  __builtin_amdgcn_s_wait_asynccnt(0);
#else
  asm volatile("s_wait_asynccnt 0x0" ::: "memory");
#endif
#endif
}

// -------- WMMA fragment loaders -------------------------------------------
// 16-bit operand layout (ISA 7.12.2): lane L (L<16) holds row M=L,
// K = k0 + {0..7} in VGPRs 0-3 and K = k0 + {16..23} in VGPRs 4-7;
// lanes 16-31 hold the same rows with K offset by +8.
// Two contiguous 16-byte chunks per lane -> two b128 loads. Unguarded:
// all A-side buffers are padded to the grid's row coverage.
__device__ __forceinline__ v16bf load_frag_g(const unsigned short* __restrict__ P,
                                             int r0, int ld, int k0) {
  const int lane = threadIdx.x & 31;
  const int half = lane >> 4;
  const unsigned short* p = P + (size_t)(r0 + (lane & 15)) * ld + k0 + 8 * half;
  FragBF f;
  f.q[0] = *reinterpret_cast<const uint4*>(p);
  f.q[1] = *reinterpret_cast<const uint4*>(p + 16);
  return f.v;
}

__device__ __forceinline__ v16bf load_frag_lds(const unsigned short* sB, int n0) {
  const int lane = threadIdx.x & 31;
  const int half = lane >> 4;
  const unsigned short* p = sB + (n0 + (lane & 15)) * LDS_STRIDE + 8 * half;
  FragBF f;
  f.q[0] = *reinterpret_cast<const uint4*>(p);
  f.q[1] = *reinterpret_cast<const uint4*>(p + 16);
  return f.v;
}

__device__ __forceinline__ v8f wmma_bf16(v16bf a, v16bf b, v8f c) {
  return __builtin_amdgcn_wmma_f32_16x16x32_bf16(
      /*neg_a=*/false, a, /*neg_b=*/false, b,
      /*c_mod=*/(short)0, c, /*reuse_a=*/false, /*reuse_b=*/false);
}

// -------- GEMM: C = A(bf16, MxK) @ B(bf16, KxN; stored transposed NxK) + bias
// Workgroup: 256(M) x 64(N); wave: 32(M) x 64(N). K % 32 == 0, N % 64 == 0.
// A buffers must be readable up to ceil(M/256)*256 rows (padded by caller).
template <bool RELU, bool BF16OUT>
__global__ __launch_bounds__(256) void gemm_wmma_bf16(
    const unsigned short* __restrict__ A,    // M x K (row-major, bf16 bits)
    const unsigned short* __restrict__ Bt,   // N x K (row-major, = B^T)
    const float* __restrict__ bias,          // N (or nullptr)
    void* __restrict__ Cout,                 // M x N (float or bf16 bits)
    int M, int N, int K)
{
  __shared__ alignas(16) unsigned short sB[2][64 * LDS_STRIDE];

  const int tid  = threadIdx.x;
  const int wave = tid >> 5;
  const int lane = tid & 31;
  const int half = lane >> 4;
  const int row0 = blockIdx.y * 256 + wave * 32;
  const int col0 = blockIdx.x * 64;

  // B staging: thread copies one 16B chunk per K-step. 64 rows x 4 chunks.
  const int brow = tid >> 2;
  const int kch  = (tid & 3) * 8;
  const unsigned short* bsrc = Bt + (size_t)(col0 + brow) * K + kch;
  unsigned short* sdst0 = &sB[0][brow * LDS_STRIDE + kch];
  unsigned short* sdst1 = &sB[1][brow * LDS_STRIDE + kch];

  v8f acc[8] = {v8f{}, v8f{}, v8f{}, v8f{}, v8f{}, v8f{}, v8f{}, v8f{}};

  stage_b128(bsrc, sdst0);                   // prologue: k0 = 0 -> buf 0
  int buf = 0;
  for (int k0 = 0; k0 < K; k0 += 32) {
    async_wait0();
    __syncthreads();                         // this iteration's B tile visible
    if (k0 + 32 < K)
      stage_b128(bsrc + (k0 + 32), buf ? sdst0 : sdst1);

    const v16bf a0 = load_frag_g(A, row0,      K, k0);
    const v16bf a1 = load_frag_g(A, row0 + 16, K, k0);
#pragma unroll
    for (int j = 0; j < 4; ++j) {
      const v16bf b = load_frag_lds(&sB[buf][0], 16 * j);
      acc[j]     = wmma_bf16(a0, b, acc[j]);
      acc[4 + j] = wmma_bf16(a1, b, acc[4 + j]);
    }
    __syncthreads();                         // done reading sB[buf]
    buf ^= 1;
  }

  // Epilogue. C/D layout: VGPR r holds row (r + 8*half), col = lane&15.
  const int nloc = lane & 15;
#pragma unroll
  for (int j = 0; j < 4; ++j) {
    const int   col = col0 + 16 * j + nloc;
    const float bv  = bias ? bias[col] : 0.0f;
#pragma unroll
    for (int t = 0; t < 2; ++t) {
      AccF af; af.v = acc[j + 4 * t];
      const int rbase = row0 + 16 * t + 8 * half;
#pragma unroll
      for (int r = 0; r < 8; ++r) {
        const int row = rbase + r;
        if (row < M) {
          float v = af.f[r] + bv;
          if (RELU) v = fmaxf(v, 0.0f);
          if (BF16OUT)
            ((unsigned short*)Cout)[(size_t)row * N + col] = f2bf(v);
          else
            ((float*)Cout)[(size_t)row * N + col] = v;
        }
      }
    }
  }
}

// -------- Weight preparation ----------------------------------------------
// Wt[n*K + k] = bf16(W[k*N + n])   (transpose + convert)
__global__ void transpose_cvt(const float* __restrict__ W,
                              unsigned short* __restrict__ Wt, int K, int N) {
  int idx = blockIdx.x * blockDim.x + threadIdx.x;
  if (idx < K * N) {
    int n = idx / K, k = idx % K;
    Wt[idx] = f2bf(W[(size_t)k * N + n]);
  }
}

// W_big^T (2048 x 1024): rows 0..1535 = [W_ioux; W_iouh]^T, rows 1536..2047 =
// [W_fx; 0]^T  (one GEMM yields iou (1536) and fx (512)).
__global__ void build_wbig(const float* __restrict__ Wx,   // 512 x 1536
                           const float* __restrict__ Wh,   // 512 x 1536
                           const float* __restrict__ Wf,   // 512 x 512
                           unsigned short* __restrict__ Wt) {
  int idx = blockIdx.x * blockDim.x + threadIdx.x;
  if (idx >= 2048 * 1024) return;
  int n = idx >> 10, k = idx & 1023;
  float v;
  if (n < 1536)
    v = (k < 512) ? Wx[(size_t)k * 1536 + n] : Wh[(size_t)(k - 512) * 1536 + n];
  else
    v = (k < 512) ? Wf[(size_t)k * 512 + (n - 1536)] : 0.0f;
  Wt[idx] = f2bf(v);
}

__global__ void build_b1(const float* __restrict__ bx, const float* __restrict__ bh,
                         const float* __restrict__ bf_, float* __restrict__ b1) {
  int n = blockIdx.x * blockDim.x + threadIdx.x;
  if (n < 2048) b1[n] = (n < 1536) ? (bx[n] + bh[n]) : bf_[n - 1536];
}

__global__ void cvt_f32_bf16(const float* __restrict__ src,
                             unsigned short* __restrict__ dst, long n) {
  long i = (long)blockIdx.x * blockDim.x + threadIdx.x;
  if (i < n) dst[i] = f2bf(src[i]);
}

// -------- Level pack: A2 = [h | sum(child hidden)], CH2 = child hidden -----
__global__ __launch_bounds__(256) void pack_level(
    const unsigned short* __restrict__ hcur,   // layer input h (bf16)
    const unsigned short* __restrict__ hnxt,   // this layer's hidden (bf16)
    const int* __restrict__ child_idx,         // N x 4
    const int* __restrict__ child_mask,        // N x 4
    unsigned short* __restrict__ A2,           // rows x 1024
    unsigned short* __restrict__ CH2,          // (4*rows) x 512
    int node0, int rows)
{
  const int j = blockIdx.x;
  if (j >= rows) return;
  const int g = node0 + j;
  int ci[4]; float cm[4];
#pragma unroll
  for (int k = 0; k < 4; ++k) {
    ci[k] = child_idx[g * 4 + k];
    cm[k] = (float)child_mask[g * 4 + k];
  }
  for (int col = threadIdx.x; col < HF; col += 256) {
    A2[(size_t)j * 1024 + col] = hcur[(size_t)g * HF + col];
    float s = 0.0f;
#pragma unroll
    for (int k = 0; k < 4; ++k) {
      float hv = (cm[k] != 0.0f) ? bf2f(hnxt[(size_t)ci[k] * HF + col]) : 0.0f;
      s += hv;
      CH2[((size_t)(j * 4 + k)) * HF + col] = f2bf(hv);
    }
    A2[(size_t)j * 1024 + HF + col] = f2bf(s);
  }
}

// -------- Level combine: gates, cell update, hidden write ------------------
__global__ __launch_bounds__(256) void combine_level(
    const float* __restrict__ OUT1,   // rows x 2048: [i|o|u|fx] (+all biases)
    const float* __restrict__ FH,     // (4*rows) x 512: ch@W_fh + b_fh
    float* __restrict__ cell,         // N x 512 f32
    unsigned short* __restrict__ hnxt,// N x 512 bf16
    const int* __restrict__ child_idx,
    const int* __restrict__ child_mask,
    int node0, int rows)
{
  const int j = blockIdx.x;
  if (j >= rows) return;
  const int g = node0 + j;
  int ci[4]; float cm[4];
#pragma unroll
  for (int k = 0; k < 4; ++k) {
    ci[k] = child_idx[g * 4 + k];
    cm[k] = (float)child_mask[g * 4 + k];
  }
  const float* o1 = OUT1 + (size_t)j * 2048;
  for (int col = threadIdx.x; col < HF; col += 256) {
    float iv = sigmoidf_(o1[col]);
    float ov = sigmoidf_(o1[512 + col]);
    float uv = tanhf(o1[1024 + col]);
    float fx = o1[1536 + col];
    float c  = iv * uv;
#pragma unroll
    for (int k = 0; k < 4; ++k) {
      float f  = sigmoidf_(fx + FH[((size_t)(j * 4 + k)) * HF + col]);
      float cc = (cm[k] != 0.0f) ? cell[(size_t)ci[k] * HF + col] : 0.0f;
      c += f * cc;
    }
    cell[(size_t)g * HF + col] = c;
    hnxt[(size_t)g * HF + col] = f2bf(ov * tanhf(c));
  }
}

// -------- Tree-embedding reduction ----------------------------------------
__global__ void zero_tree(float* __restrict__ t) { t[threadIdx.x] = 0.0f; }

__global__ void reduce_tree(const float* __restrict__ node_emb,
                            float* __restrict__ tree, int n) {
  const int col = threadIdx.x;  // 256 cols
  float s = 0.0f;
  for (int r = blockIdx.x; r < n; r += gridDim.x)
    s += node_emb[(size_t)r * OUTF + col];
  atomicAdd(&tree[col], s);
}

// ---------------------------------------------------------------------------
extern "C" void kernel_launch(void* const* d_in, const int* in_sizes, int n_in,
                              void* d_out, int out_size, void* d_ws, size_t ws_size,
                              hipStream_t stream)
{
  const float* x          = (const float*)d_in[0];
  const int*   child_idx  = (const int*)d_in[6];
  const int*   child_mask = (const int*)d_in[7];
  const float* W_in   = (const float*)d_in[8];
  const float* b_in   = (const float*)d_in[9];
  const float* W_ioux = (const float*)d_in[10];
  const float* b_ioux = (const float*)d_in[11];
  const float* W_iouh = (const float*)d_in[12];
  const float* b_iouh = (const float*)d_in[13];
  const float* W_fx   = (const float*)d_in[14];
  const float* b_fx   = (const float*)d_in[15];
  const float* W_fh   = (const float*)d_in[16];
  const float* b_fh   = (const float*)d_in[17];
  const float* W_out  = (const float*)d_in[18];
  const float* b_out  = (const float*)d_in[19];

  // ---- workspace carve-out (A-side buffers padded to 256-row coverage) ----
  char* wsp = (char*)d_ws;
  auto alloc = [&](size_t bytes) -> char* {
    char* p = wsp; wsp += (bytes + 255) & ~(size_t)255; return p;
  };
  unsigned short* WinT  = (unsigned short*)alloc((size_t)512 * 512 * 2);
  unsigned short* Wbig  = (unsigned short*)alloc((size_t)2 * 2048 * 1024 * 2);
  unsigned short* WfhT  = (unsigned short*)alloc((size_t)2 * 512 * 512 * 2);
  unsigned short* WoutT = (unsigned short*)alloc((size_t)256 * 512 * 2);
  float*          B1    = (float*)alloc((size_t)2 * 2048 * 4);
  unsigned short* hA    = (unsigned short*)alloc((size_t)NPAD * HF * 2);
  unsigned short* hB    = (unsigned short*)alloc((size_t)NPAD * HF * 2);
  float*          cell  = (float*)alloc((size_t)NNODES * HF * 4);
  unsigned short* A2    = (unsigned short*)alloc((size_t)CHUNK * 1024 * 2);
  unsigned short* CH2   = (unsigned short*)alloc((size_t)CHUNK * 4 * 512 * 2);
  float*          OUT1  = (float*)alloc((size_t)CHUNK * 2048 * 4);
  float*          FHb   = (float*)alloc((size_t)CHUNK * 4 * 512 * 4);

  // ---- weight prep (bf16 + transpose; every call: deterministic) ----
  transpose_cvt<<<(512 * 512 + 255) / 256, 256, 0, stream>>>(W_in, WinT, 512, 512);
  for (int l = 0; l < 2; ++l) {
    build_wbig<<<(2048 * 1024 + 255) / 256, 256, 0, stream>>>(
        W_ioux + (size_t)l * 512 * 1536, W_iouh + (size_t)l * 512 * 1536,
        W_fx + (size_t)l * 512 * 512, Wbig + (size_t)l * 2048 * 1024);
    transpose_cvt<<<(512 * 512 + 255) / 256, 256, 0, stream>>>(
        W_fh + (size_t)l * 512 * 512, WfhT + (size_t)l * 512 * 512, 512, 512);
    build_b1<<<8, 256, 0, stream>>>(b_ioux + l * 1536, b_iouh + l * 1536,
                                    b_fx + l * 512, B1 + l * 2048);
  }
  transpose_cvt<<<(256 * 512 + 255) / 256, 256, 0, stream>>>(W_out, WoutT, 512, 256);

  // ---- h = relu(x @ W_in + b_in), stored bf16 in hA ----
  long nx = (long)NNODES * HF;
  cvt_f32_bf16<<<(int)((nx + 255) / 256), 256, 0, stream>>>(x, hB, nx);
  {
    dim3 g(512 / 64, (NNODES + 255) / 256);
    gemm_wmma_bf16<true, true><<<g, 256, 0, stream>>>(hB, WinT, b_in, hA,
                                                      NNODES, 512, 512);
  }

  // ---- Tree-LSTM layers, bottom-up level sweep ----
  static const int LS[10] = {0, 1, 5, 21, 85, 341, 1365, 5461, 21845, 87381};
  unsigned short* hcur = hA;
  unsigned short* hnxt = hB;
  for (int l = 0; l < 2; ++l) {
    for (int d = TDEPTH; d >= 0; --d) {
      const int s = LS[d], e = LS[d + 1];
      for (int base = s; base < e; base += CHUNK) {
        const int rows = (e - base < CHUNK) ? (e - base) : CHUNK;
        pack_level<<<rows, 256, 0, stream>>>(hcur, hnxt, child_idx, child_mask,
                                             A2, CH2, base, rows);
        {
          dim3 g(2048 / 64, (rows + 255) / 256);
          gemm_wmma_bf16<false, false><<<g, 256, 0, stream>>>(
              A2, Wbig + (size_t)l * 2048 * 1024, B1 + l * 2048, OUT1,
              rows, 2048, 1024);
        }
        {
          dim3 g(512 / 64, (4 * rows + 255) / 256);
          gemm_wmma_bf16<false, false><<<g, 256, 0, stream>>>(
              CH2, WfhT + (size_t)l * 512 * 512, b_fh + l * 512, FHb,
              4 * rows, 512, 512);
        }
        combine_level<<<rows, 256, 0, stream>>>(OUT1, FHb, cell, hnxt,
                                                child_idx, child_mask, base, rows);
      }
    }
    unsigned short* t = hcur; hcur = hnxt; hnxt = t;
  }

  // ---- node_emb = h @ W_out + b_out (fp32, straight into d_out) ----
  float* node_emb = (float*)d_out;
  float* tree     = node_emb + (size_t)NNODES * OUTF;
  {
    dim3 g(OUTF / 64, (NNODES + 255) / 256);
    gemm_wmma_bf16<false, false><<<g, 256, 0, stream>>>(hcur, WoutT, b_out,
                                                        node_emb, NNODES, OUTF, 512);
  }
  zero_tree<<<1, 256, 0, stream>>>(tree);
  reduce_tree<<<256, 256, 0, stream>>>(node_emb, tree, NNODES);
}
